// decoder_53764400611707
// MI455X (gfx1250) — compile-verified, matrix operand order checked by reading, not run on hardware
//
#include <hip/hip_runtime.h>
#include <hip/hip_bf16.h>
#include <math.h>

// Problem constants from the reference
constexpr int N_  = 16384;
constexpr int Z_  = 100;   // K of main GEMM, padded to 128
constexpr int KP_ = 128;   // padded K (4 x 32 bf16 WMMA steps)
constexpr int E_  = 400;
constexpr int M0_ = 4000;
constexpr int M1_ = 2000;

typedef __attribute__((ext_vector_type(16))) __bf16 v16bf;
typedef __attribute__((ext_vector_type(8)))  float  v8f;

__device__ __forceinline__ float fast_exp2(float x) {
#if __has_builtin(__builtin_amdgcn_exp2f)
    return __builtin_amdgcn_exp2f(x);      // v_exp_f32 (base-2 native)
#else
    return exp2f(x);
#endif
}
__device__ __forceinline__ float fast_log2(float x) {
#if __has_builtin(__builtin_amdgcn_logf)
    return __builtin_amdgcn_logf(x);       // v_log_f32 (log2)
#else
    return __log2f(x);
#endif
}

// ---------------------------------------------------------------------------
// Kernel 1: topic = beta @ alpha.T, K-padded to 128, rounded to bf16 and
// pre-swizzled into the per-lane V_WMMA_F32_16X16X32_BF16 B-matrix layout:
//   packed[((ct*4 + ks)*32 + lane)*16 + i]  holds  topic[k][n] with
//   n = ct*16 + (lane&15),  k = ks*32 + (i>>3)*16 + (lane>>4)*8 + (i&7)
// ---------------------------------------------------------------------------
__global__ void topic_pack_kernel(const float* __restrict__ beta,   // (Z, E)
                                  const float* __restrict__ alpha,  // (M, E)
                                  __bf16* __restrict__ packed,      // M*128 bf16
                                  int M) {
    int idx = blockIdx.x * blockDim.x + threadIdx.x;
    if (idx >= M * KP_) return;
    int i    = idx & 15;
    int lane = (idx >> 4) & 31;
    int ks   = (idx >> 9) & 3;
    int ct   = idx >> 11;
    int n    = ct * 16 + (lane & 15);
    int k    = ks * 32 + ((i >> 3) << 4) + ((lane >> 4) << 3) + (i & 7);
    float acc = 0.0f;
    if (k < Z_) {
        const float* br = beta  + (size_t)k * E_;
        const float* ar = alpha + (size_t)n * E_;
        #pragma unroll 4
        for (int e = 0; e < E_; ++e) acc = fmaf(br[e], ar[e], acc);
    }
    packed[idx] = (__bf16)acc;   // k >= Z padded with 0
}

// ---------------------------------------------------------------------------
// Kernel 2: fused  logits = theta @ topic + bias[dom]  -> log_softmax -> out
// One workgroup (8 wave32) per 16-row stripe.
//   pass 1: WMMA tiles -> PER-LANE online (max2, sum2) in base-2 domain.
//           No cross-lane traffic in the hot loop; shuffle merge + LDS wave
//           merge happen once at the end.
//   pass 2: recompute WMMA tiles (compute is cheap vs HBM), subtract
//           logsumexp, single coalesced HBM write.
// ---------------------------------------------------------------------------
__global__ __launch_bounds__(256)
void fused_gemm_logsoftmax_kernel(const float* __restrict__ theta,  // (N, Z)
                                  const __bf16* __restrict__ Bp,    // packed topic
                                  const float* __restrict__ bias,   // (D, M)
                                  const int* __restrict__ dom,      // (N,)
                                  float* __restrict__ out,          // (N, M)
                                  int M) {
    constexpr float LOG2E = 1.44269504088896340736f;
    constexpr float LN2   = 0.69314718055994530942f;

    const int row0   = blockIdx.x * 16;
    const int lane   = threadIdx.x & 31;
    const int wave   = threadIdx.x >> 5;          // 0..7
    const int half   = lane >> 4;                 // 0 or 1
    const int nloc   = lane & 15;
    const int ntiles = M >> 4;

    // --- Build bf16 A fragments for rows row0..row0+15 (K padded to 128) ---
    v16bf a[4];
    {
        const float* trow = theta + (size_t)(row0 + nloc) * Z_;
        #pragma unroll
        for (int ks = 0; ks < 4; ++ks) {
            v16bf av;
            #pragma unroll
            for (int i = 0; i < 16; ++i) {
                int k = ks * 32 + ((i >> 3) << 4) + (half << 3) + (i & 7);
                float v = (k < Z_) ? trow[k] : 0.0f;
                av[i] = (__bf16)v;
            }
            a[ks] = av;
        }
    }

    // C/D layout: lane holds row r = v + 8*half for VGPR v, column = nloc.
    const float* brow[8];
    #pragma unroll
    for (int v = 0; v < 8; ++v)
        brow[v] = bias + (size_t)dom[row0 + v + 8 * half] * M;

    // --- Pass 1: per-lane online max/sum (base-2) over this wave's tiles ---
    float rmax[8], rsum[8];
    #pragma unroll
    for (int v = 0; v < 8; ++v) { rmax[v] = -INFINITY; rsum[v] = 0.0f; }

    for (int ct = wave; ct < ntiles; ct += 8) {          // wave-uniform bound
        v8f c = {};
        const v16bf* bptr = (const v16bf*)Bp + (size_t)(ct * 4) * 32 + lane;
        #pragma unroll
        for (int ks = 0; ks < 4; ++ks) {
            v16bf b = bptr[(size_t)ks * 32];
            c = __builtin_amdgcn_wmma_f32_16x16x32_bf16(
                    false, a[ks], false, b, (short)0, c, false, false);
        }
        int ncol = ct * 16 + nloc;
        #pragma unroll
        for (int v = 0; v < 8; ++v) {
            float l2   = (c[v] + brow[v][ncol]) * LOG2E;   // base-2 logit
            float nmax = fmaxf(rmax[v], l2);
            rsum[v] = fmaf(rsum[v], fast_exp2(rmax[v] - nmax),
                           fast_exp2(l2 - nmax));
            rmax[v] = nmax;
        }
    }

    // --- One-time merge across the 16 lanes of this half ---
    #pragma unroll
    for (int off = 1; off < 16; off <<= 1) {
        #pragma unroll
        for (int v = 0; v < 8; ++v) {
            float omax = __shfl_xor(rmax[v], off, 32);
            float osum = __shfl_xor(rsum[v], off, 32);
            float nmax = fmaxf(rmax[v], omax);
            rsum[v] = fmaf(rsum[v], fast_exp2(rmax[v] - nmax),
                           osum * fast_exp2(omax - nmax));
            rmax[v] = nmax;
        }
    }

    // --- Merge across the 8 waves via (tiny) LDS ---
    __shared__ float smax[8][16];
    __shared__ float ssum[8][16];
    if (nloc == 0) {                                      // lanes 0 and 16
        #pragma unroll
        for (int v = 0; v < 8; ++v) {
            smax[wave][v + 8 * half] = rmax[v];
            ssum[wave][v + 8 * half] = rsum[v];
        }
    }
    __syncthreads();

    float flse[8];                                        // natural logsumexp
    #pragma unroll
    for (int v = 0; v < 8; ++v) {
        int r = v + 8 * half;
        float mx = smax[0][r];
        #pragma unroll
        for (int w = 1; w < 8; ++w) mx = fmaxf(mx, smax[w][r]);
        float s = 0.0f;
        #pragma unroll
        for (int w = 0; w < 8; ++w) s += ssum[w][r] * fast_exp2(smax[w][r] - mx);
        flse[v] = LN2 * (mx + fast_log2(s));
    }

    // --- Pass 2: recompute tiles (cheap), normalize, single HBM write ---
    for (int ct = wave; ct < ntiles; ct += 8) {
        v8f c = {};
        const v16bf* bptr = (const v16bf*)Bp + (size_t)(ct * 4) * 32 + lane;
        #pragma unroll
        for (int ks = 0; ks < 4; ++ks) {
            v16bf b = bptr[(size_t)ks * 32];
            c = __builtin_amdgcn_wmma_f32_16x16x32_bf16(
                    false, a[ks], false, b, (short)0, c, false, false);
        }
        int ncol = ct * 16 + nloc;
        #pragma unroll
        for (int v = 0; v < 8; ++v) {
            int r = v + 8 * half;
            out[(size_t)(row0 + r) * M + ncol] = c[v] + brow[v][ncol] - flse[v];
        }
    }
}

// ---------------------------------------------------------------------------
extern "C" void kernel_launch(void* const* d_in, const int* in_sizes, int n_in,
                              void* d_out, int out_size, void* d_ws, size_t ws_size,
                              hipStream_t stream) {
    const float* theta  = (const float*)d_in[0];   // (N, Z)
    const float* alpha0 = (const float*)d_in[1];   // (M0, E)
    const float* alpha1 = (const float*)d_in[2];   // (M1, E)
    const float* beta   = (const float*)d_in[3];   // (Z, E)
    const float* bias0  = (const float*)d_in[4];   // (D, M0)
    const float* bias1  = (const float*)d_in[5];   // (D, M1)
    const int*   domi   = (const int*)d_in[6];     // (N,)
    float* out0 = (float*)d_out;                                  // (N, M0)
    float* out1 = (float*)d_out + (size_t)N_ * M0_;               // (N, M1)

    // Workspace: packed bf16 topic matrices (L2-resident, ~1.5 MB total)
    __bf16* packed0 = (__bf16*)d_ws;                              // M0*128
    __bf16* packed1 = packed0 + (size_t)M0_ * KP_;                // M1*128

    // Topic GEMMs + swizzle-pack (tiny: 0.5 GFLOP, reads stay in L2)
    {
        int tot0 = M0_ * KP_;
        int tot1 = M1_ * KP_;
        topic_pack_kernel<<<(tot0 + 255) / 256, 256, 0, stream>>>(beta, alpha0, packed0, M0_);
        topic_pack_kernel<<<(tot1 + 255) / 256, 256, 0, stream>>>(beta, alpha1, packed1, M1_);
    }

    // Fused WMMA GEMM + bias-gather + online log-softmax, one 16-row stripe
    // per workgroup (N/16 = 1024 workgroups, 8 wave32 each).
    fused_gemm_logsoftmax_kernel<<<N_ / 16, 256, 0, stream>>>(
        theta, packed0, bias0, domi, out0, M0_);
    fused_gemm_logsoftmax_kernel<<<N_ / 16, 256, 0, stream>>>(
        theta, packed1, bias1, domi, out1, M1_);
}
